// QAGNN_Message_Passing_77695958385208
// MI455X (gfx1250) — compile-verified, matrix operand order checked by reading, not run on hardware
//
#include <hip/hip_runtime.h>

// ---------------- problem constants (from reference) ----------------
#define DD     200        // D
#define HH     4          // heads
#define DHH    50         // D/H
#define NTE    39         // N_ETYPE + 1
#define NBATCH 256        // B graphs
#define LGR    1024       // edges per graph
#define NEDGE  262144     // E
#define NNODE  51200      // N
#define NROWS  313344     // E + N  (divisible by 128)
#define NCOMBO 628        // 39*16 edge combos + 4 node combos

#define LDSTR  40         // LDS row stride in halves (80B: bank-conflict-free, 16B aligned)

typedef __attribute__((ext_vector_type(16))) _Float16 v16h;
typedef __attribute__((ext_vector_type(8)))  _Float16 v8h;
typedef __attribute__((ext_vector_type(8)))  float    v8f;

__device__ __forceinline__ v8f wmma16(v16h a, v16h b, v8f c) {
  // D = A(16x32 f16) * B(32x16 f16) + C(16x16 f32)
  return __builtin_amdgcn_wmma_f32_16x16x32_f16(false, a, false, b, (short)0, c,
                                                false, false);
}

__device__ __forceinline__ void atomicMaxF(float* addr, float val) {
  unsigned int* ua = (unsigned int*)addr;
  unsigned int old = *ua;
  while (true) {
    float cur = __uint_as_float(old);
    if (cur >= val) break;
    unsigned int assumed = old;
    old = atomicCAS(ua, assumed, __float_as_uint(val));
    if (old == assumed) break;
  }
}

__device__ __forceinline__ v8h cvt8(float4 f0, float4 f1) {
  v8h h;
  h[0] = (_Float16)f0.x; h[1] = (_Float16)f0.y; h[2] = (_Float16)f0.z; h[3] = (_Float16)f0.w;
  h[4] = (_Float16)f1.x; h[5] = (_Float16)f1.y; h[6] = (_Float16)f1.z; h[7] = (_Float16)f1.w;
  return h;
}

// ---------------- generic fills ----------------
__global__ void fillf_k(float* __restrict__ p, float v, size_t n) {
  size_t i = (size_t)blockIdx.x * blockDim.x + threadIdx.x;
  if (i < n) p[i] = v;
}
__global__ void filli_k(int* __restrict__ p, int v, size_t n) {
  size_t i = (size_t)blockIdx.x * blockDim.x + threadIdx.x;
  if (i < n) p[i] = v;
}

// ---------------- per-edge-type mean of x[dst]-x[src] ----------------
__global__ void edge_stats_k(const float* __restrict__ x,
                             const int* __restrict__ head,
                             const int* __restrict__ tail,
                             const int* __restrict__ etype,
                             float* __restrict__ tsum, float* __restrict__ tcnt) {
  size_t i = (size_t)blockIdx.x * blockDim.x + threadIdx.x;
  size_t total = (size_t)NEDGE * DD;
  if (i >= total) return;
  int e = (int)(i / DD), d = (int)(i % DD);
  int t = etype[e];
  float v = x[(size_t)tail[e] * DD + d] - x[(size_t)head[e] * DD + d];
  atomicAdd(&tsum[t * DD + d], v);
  if (d == 0) atomicAdd(&tcnt[t], 1.0f);
}

__global__ void type_mean_k(const float* __restrict__ tsum,
                            const float* __restrict__ tcnt,
                            float* __restrict__ tmean) {
  int i = blockIdx.x * blockDim.x + threadIdx.x;
  if (i < NTE * DD) tmean[i] = tsum[i] / fmaxf(tcnt[i / DD], 1.0f);
}

// ---------------- one-hot combo counts for exact BN1 stats ----------------
__global__ void combo_count_k(const int* __restrict__ etype,
                              const int* __restrict__ head,
                              const int* __restrict__ tail,
                              const int* __restrict__ ntype,
                              int* __restrict__ ccnt) {
  size_t r = (size_t)blockIdx.x * blockDim.x + threadIdx.x;
  if (r >= NROWS) return;
  int c;
  if (r < NEDGE) c = etype[r] * 16 + ntype[head[r]] * 4 + ntype[tail[r]];
  else           c = 624 + ntype[r - NEDGE];
  atomicAdd(&ccnt[c], 1);
}

// exact BN statistics over all NROWS rows, derived from combo counts
__global__ void bn1_stats_k(const float* __restrict__ We1, const float* __restrict__ be1,
                            const float* __restrict__ g, const float* __restrict__ b,
                            const int* __restrict__ ccnt,
                            float* __restrict__ bsc, float* __restrict__ bsh) {
  int d = blockIdx.x * blockDim.x + threadIdx.x;
  if (d >= DD) return;
  double sum = 0.0, sq = 0.0;
  for (int c = 0; c < NCOMBO; ++c) {
    int n = ccnt[c];
    if (!n) continue;
    float h;
    if (c < 624) {
      int te = c >> 4, ts = (c >> 2) & 3, td = c & 3;
      h = We1[te * DD + d] + We1[(39 + ts) * DD + d] + We1[(43 + td) * DD + d] + be1[d];
    } else {
      int t = c - 624;
      h = We1[38 * DD + d] + We1[(39 + t) * DD + d] + We1[(43 + t) * DD + d] + be1[d];
    }
    sum += (double)n * h;
    sq  += (double)n * h * (double)h;
  }
  float mean = (float)(sum / (double)NROWS);
  float var  = (float)(sq  / (double)NROWS) - mean * mean;
  float sc = g[d] * rsqrtf(var + 1e-5f);
  bsc[d] = sc;
  bsh[d] = b[d] - mean * sc;
}

// h1 = relu(bn(feat @ We1 + be1))  (feat is 3-way one-hot: sum of 3 We1 rows)
__global__ void h1_apply_k(const float* __restrict__ We1, const float* __restrict__ be1,
                           const int* __restrict__ etype, const int* __restrict__ head,
                           const int* __restrict__ tail, const int* __restrict__ ntype,
                           const float* __restrict__ bsc, const float* __restrict__ bsh,
                           float* __restrict__ out) {
  size_t i = (size_t)blockIdx.x * blockDim.x + threadIdx.x;
  size_t total = (size_t)NROWS * DD;
  if (i >= total) return;
  size_t r = i / DD;
  int d = (int)(i % DD);
  int ta, tb, tc;
  if (r < NEDGE) { ta = etype[r]; tb = 39 + ntype[head[r]]; tc = 43 + ntype[tail[r]]; }
  else           { int t = ntype[r - NEDGE]; ta = 38; tb = 39 + t; tc = 43 + t; }
  float h = We1[ta * DD + d] + We1[tb * DD + d] + We1[tc * DD + d] + be1[d];
  out[i] = fmaxf(h * bsc[d] + bsh[d], 0.0f);
}

// ======================================================================
// LDS-tiled WMMA GEMM: block = 256 thr = 8 waves, C tile = 128(M) x 32(N).
// Wave w owns a 16-row strip and both 16-col subtiles (A fragment reused):
// per k-step: 6x ds_load_b128 + 2x v_wmma between one barrier pair.
// Staging chunks are 8-aligned (A) / 4-aligned (B); K,N are multiples of
// 8/4, so chunks are ALL-valid or ALL-invalid -> no scalar fallback path.
// C = relu?( (A @ W + bias) * scale )
// ======================================================================
__global__ void gemm_wmma_k(const float* __restrict__ A, int lda,
                            const float* __restrict__ Wt, int ldw,
                            const float* __restrict__ bias,
                            float* __restrict__ C, int ldc,
                            int M, int N, int K, int relu, float scale) {
  __shared__ _Float16 ldsA[128 * LDSTR];
  __shared__ _Float16 ldsB[32 * LDSTR];
  const int tid  = threadIdx.x;
  const int lane = tid & 31;
  const int wave = tid >> 5;
  const int mbase = blockIdx.y * 128;
  const int nbase = blockIdx.x * 32;
  const int wm = wave * 16;           // wave M strip in block tile
  const int hi = lane >> 4;
  const int l16 = lane & 15;

  // A staging: thread -> (row, 16-wide k chunk as two 8-chunks)
  const int am  = tid >> 1;           // 0..127
  const int akq = (tid & 1) * 16;     // 0 or 16
  const int agm = mbase + am;
  const bool arowok = agm < M;
  const float* asrc = A + (size_t)(arowok ? agm : 0) * lda;
  // B staging: thread -> (k, 4-wide n chunk)
  const int bkk = tid >> 3;           // 0..31
  const int bnq = (tid & 7) * 4;      // 0..28
  const bool bcolok = (nbase + bnq + 4) <= N;   // all-or-nothing (N % 4 == 0)

  v8f acc0 = {};
  v8f acc1 = {};
  for (int kt = 0; kt < K; kt += 32) {
    // ---- stage A tile (two 8-chunks, all-or-nothing since K % 8 == 0) ----
#pragma unroll
    for (int c = 0; c < 2; ++c) {
      int k0 = kt + akq + c * 8;
      v8h hv = {};
      if (arowok && (k0 + 8 <= K)) {
        __builtin_prefetch(asrc + k0 + 32, 0, 1);
        float4 f0 = *(const float4*)(asrc + k0);
        float4 f1 = *(const float4*)(asrc + k0 + 4);
        hv = cvt8(f0, f1);
      }
      *(v8h*)&ldsA[am * LDSTR + akq + c * 8] = hv;
    }
    // ---- stage B tile (transposed: LDS[n][k]; all-or-nothing) ----
    {
      int gk = kt + bkk;
      _Float16 h0 = (_Float16)0.f, h1 = h0, h2 = h0, h3 = h0;
      if ((gk < K) && bcolok) {
        float4 f = *(const float4*)&Wt[(size_t)gk * ldw + nbase + bnq];
        h0 = (_Float16)f.x; h1 = (_Float16)f.y; h2 = (_Float16)f.z; h3 = (_Float16)f.w;
      }
      ldsB[(bnq + 0) * LDSTR + bkk] = h0;
      ldsB[(bnq + 1) * LDSTR + bkk] = h1;
      ldsB[(bnq + 2) * LDSTR + bkk] = h2;
      ldsB[(bnq + 3) * LDSTR + bkk] = h3;
    }
    __syncthreads();
    // ---- fragments + WMMA (A reused for both N subtiles) ----
    {
      const v8h* Ab = (const v8h*)ldsA;
      const v8h* Bb = (const v8h*)ldsB;
      int ma = wm + l16;
      v8h a0 = Ab[5 * ma + hi];
      v8h a1 = Ab[5 * ma + hi + 2];
      v16h af = __builtin_shufflevector(a0, a1, 0,1,2,3,4,5,6,7,8,9,10,11,12,13,14,15);
      v8h b0 = Bb[5 * l16 + hi];
      v8h b1 = Bb[5 * l16 + hi + 2];
      v16h bf0 = __builtin_shufflevector(b0, b1, 0,1,2,3,4,5,6,7,8,9,10,11,12,13,14,15);
      v8h b2 = Bb[5 * (16 + l16) + hi];
      v8h b3 = Bb[5 * (16 + l16) + hi + 2];
      v16h bf1 = __builtin_shufflevector(b2, b3, 0,1,2,3,4,5,6,7,8,9,10,11,12,13,14,15);
      acc0 = wmma16(af, bf0, acc0);
      acc1 = wmma16(af, bf1, acc1);
    }
    __syncthreads();
  }
#pragma unroll
  for (int v = 0; v < 8; ++v) {
    int m = mbase + wm + v + hi * 8;
    if (m < M) {
      int n0 = nbase + l16;
      int n1 = nbase + 16 + l16;
      if (n0 < N) {
        float r = acc0[v];
        if (bias) r += bias[n0];
        r *= scale;
        if (relu) r = fmaxf(r, 0.0f);
        C[(size_t)m * ldc + n0] = r;
      }
      if (n1 < N) {
        float r = acc1[v];
        if (bias) r += bias[n1];
        r *= scale;
        if (relu) r = fmaxf(r, 0.0f);
        C[(size_t)m * ldc + n1] = r;
      }
    }
  }
}

// fused gather-GEMM over NROWS rows (M=NROWS, N=DD fixed):
//   A[r, k<200]  = x[ xidx[r] if r<E else r-E ][k]
//   A[r, k>=200] = (r<E ? upd[r>>10] : emb[r])[k-200]      (when use_attr)
__global__ void gather_gemm_wmma_k(const float* __restrict__ x,
                                   const int* __restrict__ xidx,
                                   const float* __restrict__ upd,
                                   const float* __restrict__ emb,
                                   int use_attr,
                                   const float* __restrict__ Wt,
                                   const float* __restrict__ bias,
                                   float* __restrict__ C,
                                   float scale, int relu) {
  __shared__ _Float16 ldsA[128 * LDSTR];
  __shared__ _Float16 ldsB[32 * LDSTR];
  const int tid  = threadIdx.x;
  const int lane = tid & 31;
  const int wave = tid >> 5;
  const int Ktot = use_attr ? (2 * DD) : DD;
  const int mbase = blockIdx.y * 128;
  const int nbase = blockIdx.x * 32;
  const int wm = wave * 16;
  const int hi = lane >> 4;
  const int l16 = lane & 15;

  // A staging: gather row pointers (rows always < NROWS; grid exact)
  const int am  = tid >> 1;           // 0..127
  const int akq = (tid & 1) * 16;     // 0 or 16
  const int ar  = mbase + am;
  const float* xrow;
  const float* arow = nullptr;
  if (ar < NEDGE) {
    xrow = x + (size_t)xidx[ar] * DD;
    if (use_attr) arow = upd + (size_t)(ar >> 10) * DD;
  } else {
    xrow = x + (size_t)(ar - NEDGE) * DD;
    if (use_attr) arow = emb + (size_t)ar * DD;
  }
  const int bkk = tid >> 3;
  const int bnq = (tid & 7) * 4;
  const bool bcolok = (nbase + bnq + 4) <= DD;

  v8f acc0 = {};
  v8f acc1 = {};
  for (int kt = 0; kt < Ktot; kt += 32) {
    // ---- stage A tile (8-chunks never straddle the k=200 boundary) ----
#pragma unroll
    for (int c = 0; c < 2; ++c) {
      int k0 = kt + akq + c * 8;
      v8h hv = {};
      if (k0 + 8 <= Ktot) {
        const float* src = (k0 < DD) ? (xrow + k0) : (arow + (k0 - DD));
        __builtin_prefetch(src + 32, 0, 1);
        float4 f0 = *(const float4*)(src);
        float4 f1 = *(const float4*)(src + 4);
        hv = cvt8(f0, f1);
      }
      *(v8h*)&ldsA[am * LDSTR + akq + c * 8] = hv;
    }
    // ---- stage B tile (transposed; all-or-nothing) ----
    {
      int gk = kt + bkk;
      _Float16 h0 = (_Float16)0.f, h1 = h0, h2 = h0, h3 = h0;
      if ((gk < Ktot) && bcolok) {
        float4 f = *(const float4*)&Wt[(size_t)gk * DD + nbase + bnq];
        h0 = (_Float16)f.x; h1 = (_Float16)f.y; h2 = (_Float16)f.z; h3 = (_Float16)f.w;
      }
      ldsB[(bnq + 0) * LDSTR + bkk] = h0;
      ldsB[(bnq + 1) * LDSTR + bkk] = h1;
      ldsB[(bnq + 2) * LDSTR + bkk] = h2;
      ldsB[(bnq + 3) * LDSTR + bkk] = h3;
    }
    __syncthreads();
    {
      const v8h* Ab = (const v8h*)ldsA;
      const v8h* Bb = (const v8h*)ldsB;
      int ma = wm + l16;
      v8h a0 = Ab[5 * ma + hi];
      v8h a1 = Ab[5 * ma + hi + 2];
      v16h af = __builtin_shufflevector(a0, a1, 0,1,2,3,4,5,6,7,8,9,10,11,12,13,14,15);
      v8h b0 = Bb[5 * l16 + hi];
      v8h b1 = Bb[5 * l16 + hi + 2];
      v16h bf0 = __builtin_shufflevector(b0, b1, 0,1,2,3,4,5,6,7,8,9,10,11,12,13,14,15);
      v8h b2 = Bb[5 * (16 + l16) + hi];
      v8h b3 = Bb[5 * (16 + l16) + hi + 2];
      v16h bf1 = __builtin_shufflevector(b2, b3, 0,1,2,3,4,5,6,7,8,9,10,11,12,13,14,15);
      acc0 = wmma16(af, bf0, acc0);
      acc1 = wmma16(af, bf1, acc1);
    }
    __syncthreads();
  }
#pragma unroll
  for (int v = 0; v < 8; ++v) {
    int m = mbase + wm + v + hi * 8;
    int n0 = nbase + l16;
    int n1 = nbase + 16 + l16;
    if (n0 < DD) {
      float val = acc0[v];
      if (bias) val += bias[n0];
      val *= scale;
      if (relu) val = fmaxf(val, 0.0f);
      C[(size_t)m * DD + n0] = val;
    }
    if (n1 < DD) {
      float val = acc1[v];
      if (bias) val += bias[n1];
      val *= scale;
      if (relu) val = fmaxf(val, 0.0f);
      C[(size_t)m * DD + n1] = val;
    }
  }
}

// ---------------- per-column BN stats (optional fused +proj[etype]) --------
__global__ void colstats_k(float* __restrict__ A, const float* __restrict__ proj,
                           const int* __restrict__ etype, int Mrows,
                           float* __restrict__ gsum, float* __restrict__ gsq) {
  __shared__ float ssum[DD];
  __shared__ float ssq[DD];
  if (threadIdx.x < DD) { ssum[threadIdx.x] = 0.f; ssq[threadIdx.x] = 0.f; }
  __syncthreads();
  int base = blockIdx.x * 128;
  for (int i = threadIdx.x; i < 128 * DD; i += 256) {
    int r = base + i / DD;
    if (r >= Mrows) break;
    int d = i % DD;
    float v = A[(size_t)r * DD + d];
    if (proj) {
      v += proj[etype[r] * DD + d];
      A[(size_t)r * DD + d] = v;
    }
    atomicAdd(&ssum[d], v);
    atomicAdd(&ssq[d], v * v);
  }
  __syncthreads();
  if (threadIdx.x < DD) {
    atomicAdd(&gsum[threadIdx.x], ssum[threadIdx.x]);
    atomicAdd(&gsq[threadIdx.x],  ssq[threadIdx.x]);
  }
}

__global__ void bn_fin_k(const float* __restrict__ gsum, const float* __restrict__ gsq,
                         const float* __restrict__ g, const float* __restrict__ b,
                         float invM, float* __restrict__ bsc, float* __restrict__ bsh) {
  int d = blockIdx.x * blockDim.x + threadIdx.x;
  if (d >= DD) return;
  float mean = gsum[d] * invM;
  float var  = gsq[d] * invM - mean * mean;
  float sc = g[d] * rsqrtf(var + 1e-5f);
  bsc[d] = sc;
  bsh[d] = b[d] - mean * sc;
}

__global__ void bn_apply_relu_k(float* __restrict__ A, const float* __restrict__ bsc,
                                const float* __restrict__ bsh, size_t total) {
  size_t i = (size_t)blockIdx.x * blockDim.x + threadIdx.x;
  if (i >= total) return;
  int d = (int)(i % DD);
  A[i] = fmaxf(A[i] * bsc[d] + bsh[d], 0.0f);
}

// ---------------- graph attention ----------------
__global__ void attn_scores_k(const float* __restrict__ q, const float* __restrict__ R,
                              float* __restrict__ scores) {
  int row = blockIdx.x * 8 + (threadIdx.x >> 5);
  if (row >= NEDGE) return;
  int b = row >> 10;
  int lane = threadIdx.x & 31;
  float s = 0.f;
  for (int d = lane; d < DD; d += 32)
    s += q[b * DD + d] * R[(size_t)row * DD + d];
  for (int off = 16; off > 0; off >>= 1) s += __shfl_xor(s, off, 32);
  if (lane == 0) scores[row] = s * 0.07071067811865475f;   // 1/sqrt(200)
}

__global__ void softmax_upd_k(const float* __restrict__ scores, const float* __restrict__ R,
                              float* __restrict__ upd) {
  __shared__ float sm[LGR];
  __shared__ float red[256];
  int b = blockIdx.x, t = threadIdx.x;
  float mx = -1e30f;
  for (int l = t; l < LGR; l += 256) {
    float v = scores[b * LGR + l];
    sm[l] = v;
    mx = fmaxf(mx, v);
  }
  red[t] = mx;
  __syncthreads();
  for (int s = 128; s > 0; s >>= 1) {
    if (t < s) red[t] = fmaxf(red[t], red[t + s]);
    __syncthreads();
  }
  mx = red[0];
  __syncthreads();
  float sum = 0.f;
  for (int l = t; l < LGR; l += 256) {
    float e = __expf(sm[l] - mx);
    sm[l] = e;
    sum += e;
  }
  red[t] = sum;
  __syncthreads();
  for (int s = 128; s > 0; s >>= 1) {
    if (t < s) red[t] += red[t + s];
    __syncthreads();
  }
  float inv = 1.0f / red[0];
  if (t < DD) {
    float acc = 0.f;
    for (int l = 0; l < LGR; ++l)
      acc += sm[l] * R[((size_t)b * LGR + l) * DD + t];
    upd[b * DD + t] = acc * inv;
  }
}

// ---------------- per-node segment softmax + aggregation ----------------
__global__ void sc_k(const float* __restrict__ qry, const float* __restrict__ key,
                     float* __restrict__ sc) {
  size_t i = (size_t)blockIdx.x * blockDim.x + threadIdx.x;
  if (i >= (size_t)NROWS * HH) return;
  size_t r = i >> 2;
  int h = (int)(i & 3);
  const float* qp = qry + r * DD + h * DHH;
  const float* kp = key + r * DD + h * DHH;
  float s = 0.f;
  for (int j = 0; j < DHH; ++j) s += qp[j] * kp[j];
  sc[i] = s;
}

__global__ void segmax_k(const float* __restrict__ sc, const int* __restrict__ head,
                         float* __restrict__ smax) {
  size_t i = (size_t)blockIdx.x * blockDim.x + threadIdx.x;
  if (i >= (size_t)NROWS * HH) return;
  size_t r = i >> 2;
  int h = (int)(i & 3);
  int seg = (r < NEDGE) ? head[r] : (int)(r - NEDGE);
  atomicMaxF(&smax[seg * HH + h], sc[i]);
}

__global__ void segexp_k(float* __restrict__ sc, const int* __restrict__ head,
                         const float* __restrict__ smax, float* __restrict__ den,
                         float* __restrict__ cnt) {
  size_t i = (size_t)blockIdx.x * blockDim.x + threadIdx.x;
  if (i >= (size_t)NROWS * HH) return;
  size_t r = i >> 2;
  int h = (int)(i & 3);
  int seg = (r < NEDGE) ? head[r] : (int)(r - NEDGE);
  float e = __expf(sc[i] - smax[seg * HH + h]);
  sc[i] = e;
  atomicAdd(&den[seg * HH + h], e);
  if (h == 0) atomicAdd(&cnt[seg], 1.0f);
}

__global__ void aggregate_k(const float* __restrict__ msg, const float* __restrict__ ex,
                            const float* __restrict__ den, const float* __restrict__ cnt,
                            const int* __restrict__ head, const int* __restrict__ tail,
                            float* __restrict__ aggr) {
  size_t i = (size_t)blockIdx.x * blockDim.x + threadIdx.x;
  if (i >= (size_t)NROWS * DD) return;
  size_t r = i / DD;
  int d = (int)(i % DD);
  int h = d / DHH;
  int seg, dst;
  if (r < NEDGE) { seg = head[r]; dst = tail[r]; }
  else           { seg = (int)(r - NEDGE); dst = seg; }
  float alpha = ex[r * HH + h] / den[seg * HH + h] * cnt[seg];
  atomicAdd(&aggr[(size_t)dst * DD + d], msg[i] * alpha);
}

// ---------------- launch ----------------
extern "C" void kernel_launch(void* const* d_in, const int* in_sizes, int n_in,
                              void* d_out, int out_size, void* d_ws, size_t ws_size,
                              hipStream_t stream) {
  (void)in_sizes; (void)n_in; (void)out_size; (void)ws_size;
  const float* x     = (const float*)d_in[0];
  const float* sent  = (const float*)d_in[2];
  const float* We1   = (const float*)d_in[3];
  const float* be1   = (const float*)d_in[4];
  const float* ge1   = (const float*)d_in[5];
  const float* bte1  = (const float*)d_in[6];
  const float* We2   = (const float*)d_in[7];
  const float* be2   = (const float*)d_in[8];
  const float* Wa1   = (const float*)d_in[9];
  const float* ba1   = (const float*)d_in[10];
  const float* ga1   = (const float*)d_in[11];
  const float* bta1  = (const float*)d_in[12];
  const float* Wa2   = (const float*)d_in[13];
  const float* ba2   = (const float*)d_in[14];
  const float* Watt  = (const float*)d_in[15];
  const float* Wk    = (const float*)d_in[16];
  const float* bk    = (const float*)d_in[17];
  const float* Wm    = (const float*)d_in[18];
  const float* bm    = (const float*)d_in[19];
  const float* Wq    = (const float*)d_in[20];
  const float* bq    = (const float*)d_in[21];
  const float* Wo1   = (const float*)d_in[22];
  const float* bo1   = (const float*)d_in[23];
  const float* go1   = (const float*)d_in[24];
  const float* bto1  = (const float*)d_in[25];
  const float* Wo2   = (const float*)d_in[26];
  const float* bo2   = (const float*)d_in[27];
  const int* eindex  = (const int*)d_in[28];
  const int* head    = eindex;            // edge_index[0] = src
  const int* tail    = eindex + NEDGE;    // edge_index[1] = dst
  const int* etype   = (const int*)d_in[29];
  const int* ntype   = (const int*)d_in[30];
  float* out = (float*)d_out;

  // workspace carve-up (floats)
  float* W0 = (float*)d_ws;
  size_t o = 0;
  float* buf1 = W0 + o; o += (size_t)NROWS * DD;   // h1 -> R -> qry -> t1
  float* buf2 = W0 + o; o += (size_t)NROWS * DD;   // emb (emb_e rows 0..E, emb_self rows E..)
  float* buf3 = W0 + o; o += (size_t)NROWS * DD;   // h2 -> key -> msg
  float* tsum = W0 + o; o += NTE * DD;
  float* tcnt = W0 + o; o += 64;
  float* tmean= W0 + o; o += NTE * DD;
  float* proj = W0 + o; o += NTE * DD;
  int*   ccnt = (int*)(W0 + o); o += NCOMBO + 4;
  float* gsum = W0 + o; o += 256;
  float* gsq  = W0 + o; o += 256;
  float* bsc  = W0 + o; o += 256;
  float* bsh  = W0 + o; o += 256;
  float* qatt = W0 + o; o += NBATCH * DD;
  float* scor = W0 + o; o += (size_t)NBATCH * LGR;
  float* updv = W0 + o; o += NBATCH * DD;
  float* scv  = W0 + o; o += (size_t)NROWS * HH;
  float* smax = W0 + o; o += (size_t)NNODE * HH;
  float* denv = W0 + o; o += (size_t)NNODE * HH;
  float* cntv = W0 + o; o += NNODE;
  float* aggr = W0 + o; o += (size_t)NNODE * DD;

  auto nb = [](size_t n) { return (unsigned)((n + 255) / 256); };
  auto ggrid = [](int M, int N) { return dim3((unsigned)((N + 31) / 32),
                                              (unsigned)((M + 127) / 128)); };
  const dim3 GG = ggrid(NROWS, DD);   // gather-GEMM grid (7 x 2448)

  // 1) per-edge-type means of x_fact
  fillf_k<<<nb(NTE * DD), 256, 0, stream>>>(tsum, 0.f, NTE * DD);
  fillf_k<<<nb(NTE), 256, 0, stream>>>(tcnt, 0.f, NTE);
  edge_stats_k<<<nb((size_t)NEDGE * DD), 256, 0, stream>>>(x, head, tail, etype, tsum, tcnt);
  type_mean_k<<<nb(NTE * DD), 256, 0, stream>>>(tsum, tcnt, tmean);

  // 2) layer-1 embedding with exact BN from combo counts
  filli_k<<<nb(NCOMBO), 256, 0, stream>>>(ccnt, 0, NCOMBO);
  combo_count_k<<<nb(NROWS), 256, 0, stream>>>(etype, head, tail, ntype, ccnt);
  bn1_stats_k<<<1, 256, 0, stream>>>(We1, be1, ge1, bte1, ccnt, bsc, bsh);
  h1_apply_k<<<nb((size_t)NROWS * DD), 256, 0, stream>>>(We1, be1, etype, head, tail,
                                                         ntype, bsc, bsh, buf1);
  // emb = h1 @ We2 + be2
  gemm_wmma_k<<<ggrid(NROWS, DD), 256, 0, stream>>>(buf1, DD, We2, DD, be2,
                                                    buf2, DD, NROWS, DD, DD, 0, 1.0f);

  // 3) h2 = BN(emb_e @ Wa1_top + mean[etype] @ Wa1_bot + ba1); R = relu(h2) @ Wa2
  gemm_wmma_k<<<ggrid(NEDGE, DD), 256, 0, stream>>>(buf2, DD, Wa1, DD, ba1,
                                                    buf3, DD, NEDGE, DD, DD, 0, 1.0f);
  gemm_wmma_k<<<ggrid(NTE, DD), 256, 0, stream>>>(tmean, DD, Wa1 + DD * DD, DD, nullptr,
                                                  proj, DD, NTE, DD, DD, 0, 1.0f);
  fillf_k<<<nb(512), 256, 0, stream>>>(gsum, 0.f, 512);   // gsum+gsq contiguous
  colstats_k<<<NEDGE / 128, 256, 0, stream>>>(buf3, proj, etype, NEDGE, gsum, gsq);
  bn_fin_k<<<1, 256, 0, stream>>>(gsum, gsq, ga1, bta1, 1.0f / (float)NEDGE, bsc, bsh);
  bn_apply_relu_k<<<nb((size_t)NEDGE * DD), 256, 0, stream>>>(buf3, bsc, bsh, (size_t)NEDGE * DD);
  gemm_wmma_k<<<ggrid(NEDGE, DD), 256, 0, stream>>>(buf3, DD, Wa2, DD, ba2,
                                                    buf1, DD, NEDGE, DD, DD, 0, 1.0f);

  // 4) graph attention: q = sent @ Watt, softmax over 1024 edges, upd per graph
  gemm_wmma_k<<<ggrid(NBATCH, DD), 256, 0, stream>>>(sent, DD, Watt, DD, nullptr,
                                                     qatt, DD, NBATCH, DD, DD, 0, 1.0f);
  attn_scores_k<<<NEDGE / 8, 256, 0, stream>>>(qatt, buf1, scor);
  softmax_upd_k<<<NBATCH, 256, 0, stream>>>(scor, buf1, updv);

  // 5) key / qry / sc  (gather-fused GEMMs, K=400 concat for key)
  gather_gemm_wmma_k<<<GG, 256, 0, stream>>>(x, tail, updv, buf2, 1, Wk, bk,
                                             buf3, 1.0f, 0);                  // key
  gather_gemm_wmma_k<<<GG, 256, 0, stream>>>(x, head, nullptr, buf2, 0, Wq, bq,
                                             buf1, 0.1414213562373095f, 0);   // qry/sqrt(DH)
  sc_k<<<nb((size_t)NROWS * HH), 256, 0, stream>>>(buf1, buf3, scv);

  // 6) segment softmax over source nodes (scaled by segment count)
  fillf_k<<<nb((size_t)NNODE * HH), 256, 0, stream>>>(smax, -1e30f, (size_t)NNODE * HH);
  segmax_k<<<nb((size_t)NROWS * HH), 256, 0, stream>>>(scv, head, smax);
  fillf_k<<<nb((size_t)NNODE * HH), 256, 0, stream>>>(denv, 0.f, (size_t)NNODE * HH);
  fillf_k<<<nb(NNODE), 256, 0, stream>>>(cntv, 0.f, NNODE);
  segexp_k<<<nb((size_t)NROWS * HH), 256, 0, stream>>>(scv, head, smax, denv, cntv);

  // 7) msg and weighted scatter-add to destination nodes
  gather_gemm_wmma_k<<<GG, 256, 0, stream>>>(x, head, updv, buf2, 1, Wm, bm,
                                             buf3, 1.0f, 0);                  // msg
  fillf_k<<<nb((size_t)NNODE * DD), 256, 0, stream>>>(aggr, 0.f, (size_t)NNODE * DD);
  aggregate_k<<<nb((size_t)NROWS * DD), 256, 0, stream>>>(buf3, scv, denv, cntv,
                                                          head, tail, aggr);

  // 8) output head: relu(BN(aggr @ Wo1 + bo1)) @ Wo2 + bo2
  gemm_wmma_k<<<ggrid(NNODE, DD), 256, 0, stream>>>(aggr, DD, Wo1, DD, bo1,
                                                    buf1, DD, NNODE, DD, DD, 0, 1.0f);
  fillf_k<<<nb(512), 256, 0, stream>>>(gsum, 0.f, 512);
  colstats_k<<<NNODE / 128, 256, 0, stream>>>(buf1, nullptr, nullptr, NNODE, gsum, gsq);
  bn_fin_k<<<1, 256, 0, stream>>>(gsum, gsq, go1, bto1, 1.0f / (float)NNODE, bsc, bsh);
  bn_apply_relu_k<<<nb((size_t)NNODE * DD), 256, 0, stream>>>(buf1, bsc, bsh, (size_t)NNODE * DD);
  gemm_wmma_k<<<ggrid(NNODE, DD), 256, 0, stream>>>(buf1, DD, Wo2, DD, bo2,
                                                    out, DD, NNODE, DD, DD, 0, 1.0f);
}
